// EncoderBlock_21294447854259
// MI455X (gfx1250) — matrix operand order repaired and checked by reading, not compiled, not run on hardware
//
#include <hip/hip_runtime.h>
#include <hip/hip_bf16.h>

// EncoderBlock: h1 = x + MHA(LN1(x)); out = h1 + FFN(LN2(h1))
// B=4 S=2048 D=1024 H=16 DK=64 DFF=4096, fp32 in/out.
// ~275 GFLOP vs ~200MB touched -> compute bound -> bf16 WMMA everywhere.
// CDNA5 paths used: v_wmma_f32_16x16x32_bf16, tensor_load_to_lds (TDM, double
// buffered), s_wait_tensorcnt, global_prefetch_b8, ds_load_b128 operand feeds.

#define B_   4
#define S_   2048
#define D_   1024
#define H_   16
#define DK_  64
#define DFF_ 4096
#define BS_  (B_ * S_)

typedef __bf16 bf16;
typedef __attribute__((ext_vector_type(16))) __bf16 v16bf;
typedef __attribute__((ext_vector_type(8)))  float  v8f;
typedef __attribute__((ext_vector_type(4)))  unsigned int v4u;
typedef __attribute__((ext_vector_type(8)))  int v8i;
typedef __attribute__((ext_vector_type(4)))  int v4i;

// ---------------------------------------------------------------- helpers

static __device__ inline v16bf ld_pair16(const bf16* p0, const bf16* p1) {
  union { v16bf v; uint4 u[2]; } x;
  x.u[0] = *reinterpret_cast<const uint4*>(p0);
  x.u[1] = *reinterpret_cast<const uint4*>(p1);
  return x.v;
}

static __device__ inline v8f wmma_bf16(v16bf a, v16bf b, v8f c) {
  return __builtin_amdgcn_wmma_f32_16x16x32_bf16(false, a, false, b, (short)0, c,
                                                 false, false);
}

static __device__ inline float half16_max(float t) {
  t = fmaxf(t, __shfl_xor(t, 1, 32));
  t = fmaxf(t, __shfl_xor(t, 2, 32));
  t = fmaxf(t, __shfl_xor(t, 4, 32));
  t = fmaxf(t, __shfl_xor(t, 8, 32));
  return t;
}
static __device__ inline float half16_sum(float t) {
  t += __shfl_xor(t, 1, 32);
  t += __shfl_xor(t, 2, 32);
  t += __shfl_xor(t, 4, 32);
  t += __shfl_xor(t, 8, 32);
  return t;
}

// TDM 2-D tile load: tile_d0 contiguous elements per row, tile_d1 rows with
// row stride `stride` (all in 2-byte elements). D# packed per ISA 08 §8.
static __device__ inline void tdm_load_2d(unsigned lds_off, const void* gptr,
                                          unsigned tile_d0, unsigned tile_d1,
                                          unsigned long long stride) {
  const unsigned long long ga = (unsigned long long)(uintptr_t)gptr;
  const unsigned long long td0 = stride;          // tensor_dim0 (OOB bound)
  const unsigned long long td1 = 1ull << 20;      // tensor_dim1 (generous)
  v4u g0;
  g0[0] = 1u;                                        // count=1, user mode
  g0[1] = lds_off;                                   // lds_addr (bytes)
  g0[2] = (unsigned)(ga & 0xffffffffu);              // global_addr lo
  g0[3] = (unsigned)((ga >> 32) & 0x01ffffffu) | (2u << 30);  // addr hi | type=2
  v8i g1;
  g1[0] = (int)(1u << 16);                           // wg_mask=0, data_size=2B
  g1[1] = (int)((td0 & 0xffffu) << 16);              // bar_addr=0 | tdim0[15:0]
  g1[2] = (int)(((td0 >> 16) & 0xffffu) | ((td1 & 0xffffu) << 16));
  g1[3] = (int)(((td1 >> 16) & 0xffffu) | (tile_d0 << 16));
  g1[4] = (int)(tile_d1 & 0xffffu);                  // tile_dim1 | tile_dim2=0
  g1[5] = (int)(stride & 0xffffffffu);               // dim0_stride lo
  g1[6] = (int)((stride >> 32) & 0xffffu);           // dim0_stride hi | d1s lo=0
  g1[7] = 0;
  const v4i z = {0, 0, 0, 0};
#if defined(__clang_major__) && (__clang_major__ >= 23)
  const v8i z8 = {0, 0, 0, 0, 0, 0, 0, 0};
  __builtin_amdgcn_tensor_load_to_lds(g0, g1, z, z, z8, 0);
#else
  __builtin_amdgcn_tensor_load_to_lds(g0, g1, z, z, 0);
#endif
}

// ---------------------------------------------------------------- cvt fp32 -> bf16

__global__ __launch_bounds__(256) void cvt_f32_bf16(const float* __restrict__ in,
                                                    bf16* __restrict__ out, int n) {
  int i = blockIdx.x * 256 + threadIdx.x;
  if (i < n) out[i] = (bf16)in[i];
}

// ---------------------------------------------------------------- LayerNorm (ddof=1)

__global__ __launch_bounds__(256) void layernorm_bf16_kernel(
    const float* __restrict__ x, const float* __restrict__ gamma,
    const float* __restrict__ beta, bf16* __restrict__ y) {
  __shared__ float part[8];
  const int row  = blockIdx.x;
  const int t    = threadIdx.x;
  const int lane = t & 31;
  const int wave = t >> 5;

  const float4 v = reinterpret_cast<const float4*>(x + (size_t)row * D_)[t];

  float s = v.x + v.y + v.z + v.w;
  for (int m = 1; m < 32; m <<= 1) s += __shfl_xor(s, m, 32);
  if (lane == 0) part[wave] = s;
  __syncthreads();
  float tot = 0.f;
#pragma unroll
  for (int i = 0; i < 8; ++i) tot += part[i];
  const float mean = tot * (1.0f / D_);

  const float dx = v.x - mean, dy = v.y - mean, dz = v.z - mean, dw = v.w - mean;
  float sq = dx * dx + dy * dy + dz * dz + dw * dw;
  for (int m = 1; m < 32; m <<= 1) sq += __shfl_xor(sq, m, 32);
  __syncthreads();
  if (lane == 0) part[wave] = sq;
  __syncthreads();
  float tot2 = 0.f;
#pragma unroll
  for (int i = 0; i < 8; ++i) tot2 += part[i];
  const float inv = rsqrtf(tot2 * (1.0f / (D_ - 1)) + 1e-9f);

  const int c = t * 4;
  bf16* yr = y + (size_t)row * D_ + c;
  yr[0] = (bf16)(gamma[c + 0] * dx * inv + beta[c + 0]);
  yr[1] = (bf16)(gamma[c + 1] * dy * inv + beta[c + 1]);
  yr[2] = (bf16)(gamma[c + 2] * dz * inv + beta[c + 2]);
  yr[3] = (bf16)(gamma[c + 3] * dw * inv + beta[c + 3]);
}

// ---------------------------------------------------------------- V transpose
// vt[(b*H+h)*64 + d][s] = v[b*S+s][h*64+d]; LDS-tiled 64x64 so both global
// sides stay 32-byte-contiguous per lane.

__global__ __launch_bounds__(256) void transpose_v_kernel(
    const bf16* __restrict__ v, bf16* __restrict__ vt) {
  __shared__ bf16 tile[64][68];  // padded rows to spread LDS banks
  const int bh = blockIdx.y;     // b*H + h
  const int b  = bh >> 4, h = bh & 15;
  const int s0 = blockIdx.x * 64;
  const int t  = threadIdx.x;
  const int r  = t >> 2, seg = t & 3;

  const bf16* src = v + (size_t)(b * S_ + s0 + r) * D_ + h * 64 + seg * 16;
  union { uint4 u; bf16 e[8]; } a0, a1;
  a0.u = reinterpret_cast<const uint4*>(src)[0];
  a1.u = reinterpret_cast<const uint4*>(src)[1];
#pragma unroll
  for (int j = 0; j < 8; ++j) tile[r][seg * 16 + j] = a0.e[j];
#pragma unroll
  for (int j = 0; j < 8; ++j) tile[r][seg * 16 + 8 + j] = a1.e[j];
  __syncthreads();

  const int dr = t >> 2, ks = (t & 3) * 16;
  union { uint4 u; bf16 e[8]; } o0, o1;
#pragma unroll
  for (int j = 0; j < 8; ++j) o0.e[j] = tile[ks + j][dr];
#pragma unroll
  for (int j = 0; j < 8; ++j) o1.e[j] = tile[ks + 8 + j][dr];
  bf16* dst = vt + ((size_t)bh * 64 + dr) * S_ + s0 + ks;
  reinterpret_cast<uint4*>(dst)[0] = o0.u;
  reinterpret_cast<uint4*>(dst)[1] = o1.u;
}

// ---------------------------------------------------------------- WMMA GEMM
// C[M,N] = A[M,K] @ W[N,K]^T (+bias)(+relu)(+residual). 8 waves, tile 32x64/wave.

template <bool HAS_BIAS, bool RELU, bool RESIDUAL, bool OUT_BF16>
__global__ __launch_bounds__(256) void gemm_wmma_kernel(
    const bf16* __restrict__ A, const bf16* __restrict__ W,
    const float* __restrict__ bias, const float* __restrict__ resid,
    bf16* __restrict__ outB, float* __restrict__ outF, int M, int N, int K) {
  const int lane = threadIdx.x & 31;
  const int wave = threadIdx.x >> 5;
  const int l15  = lane & 15;
  const int lh   = lane >> 4;
  const int wm   = wave & 3;
  const int wn   = wave >> 2;
  const int row0 = blockIdx.y * 128 + wm * 32;
  const int col0 = blockIdx.x * 128 + wn * 64;

  v8f acc[2][4] = {};

  const bf16* aptr0 = A + (size_t)(row0 + l15) * K + lh * 8;
  const bf16* aptr1 = A + (size_t)(row0 + 16 + l15) * K + lh * 8;
  const bf16* bptr[4];
#pragma unroll
  for (int j = 0; j < 4; ++j)
    bptr[j] = W + (size_t)(col0 + j * 16 + l15) * K + lh * 8;

  for (int k0 = 0; k0 < K; k0 += 32) {
    v16bf a0 = ld_pair16(aptr0 + k0, aptr0 + k0 + 16);
    v16bf a1 = ld_pair16(aptr1 + k0, aptr1 + k0 + 16);
    __builtin_prefetch(aptr0 + k0 + 64, 0, 3);   // global_prefetch_b8
    __builtin_prefetch(bptr[0] + k0 + 64, 0, 3);
#pragma unroll
    for (int j = 0; j < 4; ++j) {
      v16bf b = ld_pair16(bptr[j] + k0, bptr[j] + k0 + 16);
      acc[0][j] = wmma_bf16(a0, b, acc[0][j]);
      acc[1][j] = wmma_bf16(a1, b, acc[1][j]);
    }
  }

#pragma unroll
  for (int i = 0; i < 2; ++i) {
#pragma unroll
    for (int j = 0; j < 4; ++j) {
      const int col = col0 + j * 16 + l15;
      const float bv = HAS_BIAS ? bias[col] : 0.0f;
#pragma unroll
      for (int r = 0; r < 8; ++r) {
        const int row = row0 + i * 16 + r + lh * 8;
        float vv = acc[i][j][r] + bv;
        if (RELU) vv = fmaxf(vv, 0.0f);
        if (RESIDUAL) vv += resid[(size_t)row * N + col];
        if (OUT_BF16) outB[(size_t)row * N + col] = (bf16)vv;
        else          outF[(size_t)row * N + col] = vv;
      }
    }
  }
}

// ---------------------------------------------------------------- Flash attention
// Grid: (S/128, H, B). 8 waves/WG, each wave a 16-row Q tile (dk=64).
// K / V^T tiles (32 keys) DMA'd by the Tensor Data Mover, double buffered:
// wave0 issues tiles for block i+1, waits tensorcnt<=2 (block i complete),
// then all waves WMMA + online softmax while TDM streams the next block.

__global__ __launch_bounds__(256) void flash_attn_kernel(
    const bf16* __restrict__ q, const bf16* __restrict__ k,
    const bf16* __restrict__ vt, const int* __restrict__ mask,
    bf16* __restrict__ out) {
  __shared__ bf16 klds[2][32 * 64];     // [key][dim]
  __shared__ bf16 vtlds[2][64 * 32];    // [dim][key]
  __shared__ bf16 plds[8][16 * 32];     // per-wave P tile [qrow][key]

  const int b    = blockIdx.z;
  const int h    = blockIdx.y;
  const int lane = threadIdx.x & 31;
  const int wave = threadIdx.x >> 5;
  const int l15  = lane & 15;
  const int lh   = lane >> 4;
  const int q0   = blockIdx.x * 128 + wave * 16;
  const int rowb = b * S_;
  const int colb = h * DK_;

  const bf16* kbase  = k + (size_t)rowb * D_ + colb;            // + j*D_ per key
  const bf16* vtbase = vt + (size_t)(b * H_ + h) * 64 * S_;     // + d*S_ + key

  // Resident Q fragments: 16x64 = two 16x32 A operands.
  v16bf qa[2];
#pragma unroll
  for (int c = 0; c < 2; ++c) {
    const bf16* p = q + (size_t)(rowb + q0 + l15) * D_ + colb + c * 32 + lh * 8;
    qa[c] = ld_pair16(p, p + 16);
  }

  v8f o[4] = {};
  float mrow[8], lrow[8];
#pragma unroll
  for (int r = 0; r < 8; ++r) { mrow[r] = -1e30f; lrow[r] = 0.0f; }

  const float scale = 0.125f;  // 1/sqrt(64)
  const int NB = S_ / 32;

  if (wave == 0) {  // prologue: DMA block 0 into buffer 0
    tdm_load_2d((unsigned)(uintptr_t)&klds[0][0], kbase, 64, 32, D_);
    tdm_load_2d((unsigned)(uintptr_t)&vtlds[0][0], vtbase, 32, 64, S_);
  }

  for (int i = 0; i < NB; ++i) {
    const int j0  = i * 32;
    const int buf = i & 1;
    if (wave == 0) {
      if (i + 1 < NB) {  // DMA next block into the other buffer, then wait
        tdm_load_2d((unsigned)(uintptr_t)&klds[buf ^ 1][0],
                    kbase + (size_t)(j0 + 32) * D_, 64, 32, D_);
        tdm_load_2d((unsigned)(uintptr_t)&vtlds[buf ^ 1][0],
                    vtbase + (j0 + 32), 32, 64, S_);
        __builtin_amdgcn_s_wait_tensorcnt((short)2);  // block i landed
      } else {
        __builtin_amdgcn_s_wait_tensorcnt((short)0);
      }
    }
    __syncthreads();  // tiles for block i visible to all waves

    const bf16* kt  = &klds[buf][0];
    const bf16* vtt = &vtlds[buf][0];

    // Scores: S[16q x 32keys] = Q(16x64) @ K(32x64)^T -> 4 WMMAs
    v8f sc[2] = {};
#pragma unroll
    for (int nt = 0; nt < 2; ++nt) {
#pragma unroll
      for (int c = 0; c < 2; ++c) {
        const bf16* p = kt + (nt * 16 + l15) * 64 + c * 32 + lh * 8;
        v16bf kb = ld_pair16(p, p + 16);
        sc[nt] = wmma_bf16(qa[c], kb, sc[nt]);
      }
    }

    const int mk0 = mask[b * S_ + j0 + l15];
    const int mk1 = mask[b * S_ + j0 + 16 + l15];

    // Online softmax (each C-layout row spans 16 lanes of one half-wave).
#pragma unroll
    for (int r = 0; r < 8; ++r) {
      float s0 = sc[0][r] * scale;
      float s1 = sc[1][r] * scale;
      if (mk0 == 0) s0 = -1e9f;
      if (mk1 == 0) s1 = -1e9f;
      const float bmax = half16_max(fmaxf(s0, s1));
      const float mn   = fmaxf(mrow[r], bmax);
      const float corr = __expf(mrow[r] - mn);
      const float p0   = __expf(s0 - mn);
      const float p1   = __expf(s1 - mn);
      lrow[r] = lrow[r] * corr + half16_sum(p0 + p1);
      mrow[r] = mn;
#pragma unroll
      for (int d = 0; d < 4; ++d) o[d][r] *= corr;
      const int prow = r + lh * 8;
      plds[wave][prow * 32 + l15]      = (bf16)p0;
      plds[wave][prow * 32 + 16 + l15] = (bf16)p1;
    }

    // Same-wave LDS store -> load ordering for the P re-swizzle.
    asm volatile("s_wait_dscnt 0" ::: "memory");

    // O(16x64) += P(16x32) @ V(32x64) -> 4 WMMAs
    const bf16* pp = &plds[wave][l15 * 32 + lh * 8];
    v16bf pa = ld_pair16(pp, pp + 16);
#pragma unroll
    for (int nt = 0; nt < 4; ++nt) {
      const bf16* vp = vtt + (nt * 16 + l15) * 32 + lh * 8;
      v16bf vb = ld_pair16(vp, vp + 16);
      o[nt] = wmma_bf16(pa, vb, o[nt]);
    }
    __syncthreads();  // all waves done reading buf before TDM reuses it
  }

  // Finalize: O /= l, write bf16 [BS, D] at head columns.
#pragma unroll
  for (int r = 0; r < 8; ++r) {
    const float inv = 1.0f / lrow[r];
    const int rowg = rowb + q0 + r + lh * 8;
#pragma unroll
    for (int d = 0; d < 4; ++d)
      out[(size_t)rowg * D_ + colb + d * 16 + l15] = (bf16)(o[d][r] * inv);
  }
}

// ---------------------------------------------------------------- launch

extern "C" void kernel_launch(void* const* d_in, const int* in_sizes, int n_in,
                              void* d_out, int out_size, void* d_ws, size_t ws_size,
                              hipStream_t stream) {
  (void)in_sizes; (void)n_in; (void)out_size; (void)ws_size;
  const float* x    = (const float*)d_in[0];
  const int*   mask = (const int*)d_in[1];
  const float* wq   = (const float*)d_in[2];
  const float* wk   = (const float*)d_in[3];
  const float* wv   = (const float*)d_in[4];
  const float* wo   = (const float*)d_in[5];
  const float* w1   = (const float*)d_in[6];
  const float* b1   = (const float*)d_in[7];
  const float* w2   = (const float*)d_in[8];
  const float* b2   = (const float*)d_in[9];
  const float* g1   = (const float*)d_in[10];
  const float* be1  = (const float*)d_in[11];
  const float* g2   = (const float*)d_in[12];
  const float* be2  = (const float*)d_in[13];
  float* outp = (float*)d_out;

  char* ws = (char*)d_ws;
  size_t off = 0;
  auto alloc = [&](size_t bytes) {
    char* p = ws + off;
    off += (bytes + 255) & ~size_t(255);
    return p;
  };
  const size_t DD = (size_t)D_ * D_;
  bf16* wq_b = (bf16*)alloc(DD * 2);
  bf16* wk_b = (bf16*)alloc(DD * 2);
  bf16* wv_b = (bf16*)alloc(DD * 2);
  bf16* wo_b = (bf16*)alloc(DD * 2);
  bf16* w1_b = (bf16*)alloc((size_t)DFF_ * D_ * 2);
  bf16* w2_b = (bf16*)alloc((size_t)D_ * DFF_ * 2);
  bf16* xn   = (bf16*)alloc((size_t)BS_ * D_ * 2);
  bf16* qb   = (bf16*)alloc((size_t)BS_ * D_ * 2);
  bf16* kb   = (bf16*)alloc((size_t)BS_ * D_ * 2);
  bf16* vb   = (bf16*)alloc((size_t)BS_ * D_ * 2);
  bf16* vtb  = (bf16*)alloc((size_t)BS_ * D_ * 2);  // B*H*64 x S
  bf16* attn = (bf16*)alloc((size_t)BS_ * D_ * 2);
  float* h1  = (float*)alloc((size_t)BS_ * D_ * 4);
  bf16* z    = (bf16*)alloc((size_t)BS_ * D_ * 2);
  bf16* mid  = (bf16*)alloc((size_t)BS_ * DFF_ * 2);

  auto cvt = [&](const float* src, bf16* dst, int n) {
    hipLaunchKernelGGL(cvt_f32_bf16, dim3((n + 255) / 256), dim3(256), 0, stream,
                       src, dst, n);
  };
  cvt(wq, wq_b, (int)DD);
  cvt(wk, wk_b, (int)DD);
  cvt(wv, wv_b, (int)DD);
  cvt(wo, wo_b, (int)DD);
  cvt(w1, w1_b, DFF_ * D_);
  cvt(w2, w2_b, D_ * DFF_);

  hipLaunchKernelGGL(layernorm_bf16_kernel, dim3(BS_), dim3(256), 0, stream,
                     x, g1, be1, xn);

  const dim3 gP(D_ / 128, BS_ / 128), blk(256);
  hipLaunchKernelGGL((gemm_wmma_kernel<false, false, false, true>), gP, blk, 0,
                     stream, xn, wq_b, nullptr, nullptr, qb, nullptr, BS_, D_, D_);
  hipLaunchKernelGGL((gemm_wmma_kernel<false, false, false, true>), gP, blk, 0,
                     stream, xn, wk_b, nullptr, nullptr, kb, nullptr, BS_, D_, D_);
  hipLaunchKernelGGL((gemm_wmma_kernel<false, false, false, true>), gP, blk, 0,
                     stream, xn, wv_b, nullptr, nullptr, vb, nullptr, BS_, D_, D_);

  hipLaunchKernelGGL(transpose_v_kernel, dim3(S_ / 64, B_ * H_), blk, 0, stream,
                     vb, vtb);

  hipLaunchKernelGGL(flash_attn_kernel, dim3(S_ / 128, H_, B_), blk, 0, stream,
                     qb, kb, vtb, mask, attn);

  hipLaunchKernelGGL((gemm_wmma_kernel<false, false, true, false>), gP, blk, 0,
                     stream, attn, wo_b, nullptr, x, nullptr, h1, BS_, D_, D_);

  hipLaunchKernelGGL(layernorm_bf16_kernel, dim3(BS_), dim3(256), 0, stream,
                     h1, g2, be2, z);

  hipLaunchKernelGGL((gemm_wmma_kernel<true, true, false, true>),
                     dim3(DFF_ / 128, BS_ / 128), blk, 0, stream,
                     z, w1_b, b1, nullptr, mid, nullptr, BS_, DFF_, D_);

  hipLaunchKernelGGL((gemm_wmma_kernel<true, false, true, false>), gP, blk, 0,
                     stream, mid, w2_b, b2, h1, nullptr, outp, BS_, D_, DFF_);
}
